// InterFusion_75307956568506
// MI455X (gfx1250) — matrix operand order repaired and checked by reading, not compiled, not run on hardware
//
#include <hip/hip_runtime.h>

typedef __bf16 bf16_t;
typedef __attribute__((ext_vector_type(16))) __bf16 v16bf;
typedef __attribute__((ext_vector_type(8)))  __bf16 v8bf;
typedef __attribute__((ext_vector_type(8)))  float  v8f;

static constexpr int Bn = 16;    // batch
static constexpr int Cn = 256;   // channels
static constexpr int Nn = 4096;  // H*W
static constexpr int KSPLIT = 2; // deterministic split-K for the correlation GEMM

// ---------------------------------------------------------------------------
// Fragment loaders. k order used here for bf16 WMMA (16x16x32), same for A and B:
//  A (16x32, row-major [M,K]):  lane L (r=L&15, sel=L>>4) holds row M=r,
//    halves 0..15 = K[k0 + sel*16 .. +15] (contiguous).
//  B (32x16, stored as [N,K] row-major): lane L holds col N=r,
//    halves 0..15 = K[k0 + sel*16 .. +15] (contiguous).
// Both are two 16-byte contiguous global_load_b128 per lane.
// ---------------------------------------------------------------------------
__device__ __forceinline__ v16bf load_frag2(const bf16_t* p, int off2) {
    v8bf lo = *(const v8bf*)(p);
    v8bf hi = *(const v8bf*)(p + off2);
    v16bf r;
#pragma unroll
    for (int i = 0; i < 8; ++i) { r[i] = lo[i]; r[i + 8] = hi[i]; }
    return r;
}

// ---------------------------------------------------------------------------
// K1: per-(b,c) L2 norm over N, emit x1n, x2n (normalized bf16) and s = x1n+x2n
// grid (Cn, Bn), block 256
// ---------------------------------------------------------------------------
__global__ void __launch_bounds__(256)
norm_rows_kernel(const float* __restrict__ x1, const float* __restrict__ x2,
                 bf16_t* __restrict__ x1n, bf16_t* __restrict__ x2n,
                 bf16_t* __restrict__ s)
{
    const int t = threadIdx.x;
    const size_t base = (size_t)(blockIdx.y * Cn + blockIdx.x) * Nn;
    float r1[16], r2[16];
    float s1 = 0.f, s2 = 0.f;
#pragma unroll
    for (int i = 0; i < 16; ++i) {
        const int idx = t + i * 256;
        r1[i] = x1[base + idx];
        r2[i] = x2[base + idx];
        s1 += r1[i] * r1[i];
        s2 += r2[i] * r2[i];
    }
    __shared__ float red1[256], red2[256];
    red1[t] = s1; red2[t] = s2;
    __syncthreads();
    for (int off = 128; off > 0; off >>= 1) {
        if (t < off) { red1[t] += red1[t + off]; red2[t] += red2[t + off]; }
        __syncthreads();
    }
    const float inv1 = 1.0f / fmaxf(sqrtf(red1[0]), 1e-12f);
    const float inv2 = 1.0f / fmaxf(sqrtf(red2[0]), 1e-12f);
#pragma unroll
    for (int i = 0; i < 16; ++i) {
        const int idx = t + i * 256;
        const float a = r1[i] * inv1;
        const float b = r2[i] * inv2;
        x1n[base + idx] = (bf16_t)a;
        x2n[base + idx] = (bf16_t)b;
        s[base + idx]   = (bf16_t)(a + b);
    }
}

// ---------------------------------------------------------------------------
// K2: bf16 transpose of the RAW inputs: xT[b][n][c] = x[b][c][n]
// grid (Nn/32, Cn/32, 2*Bn), block (32,8)
// ---------------------------------------------------------------------------
__global__ void transpose_bf16_kernel(const float* __restrict__ x1,
                                      const float* __restrict__ x2,
                                      bf16_t* __restrict__ xT1,
                                      bf16_t* __restrict__ xT2)
{
    __shared__ float tile[32][33];
    const int zi = blockIdx.z;
    const int b  = zi >> 1;
    const float* src = (zi & 1) ? x2 : x1;
    bf16_t*      dst = (zi & 1) ? xT2 : xT1;
    const size_t base = (size_t)b * Cn * Nn;
    const int n0 = blockIdx.x * 32, c0 = blockIdx.y * 32;
#pragma unroll
    for (int k = 0; k < 4; ++k)
        tile[threadIdx.y + 8 * k][threadIdx.x] =
            src[base + (size_t)(c0 + threadIdx.y + 8 * k) * Nn + n0 + threadIdx.x];
    __syncthreads();
#pragma unroll
    for (int k = 0; k < 4; ++k)
        dst[base + (size_t)(n0 + threadIdx.y + 8 * k) * Cn + c0 + threadIdx.x] =
            (bf16_t)tile[threadIdx.x][threadIdx.y + 8 * k];
}

// ---------------------------------------------------------------------------
// K3: dual correlation GEMM with 2-way split-K (deterministic partials):
//   S1[ks] = s . x1n^T (K slice),  S2[ks] = s . x2n^T   (M=N=256, K=4096)
// 64x64 output tile per 128-thread block (4 waves, 2x2), each wave computes
// 2x2 16x16 tiles for BOTH streams = 8 WMMA per k-step.
// grid (Cn/64, Cn/64, Bn*KSPLIT) = (4,4,32) -> 512 blocks.
// ---------------------------------------------------------------------------
__global__ void __launch_bounds__(128)
attn_gemm_kernel(const bf16_t* __restrict__ Asum,  // [B][256][4096]
                 const bf16_t* __restrict__ Bt1,   // x1n [B][256][4096]
                 const bf16_t* __restrict__ Bt2,   // x2n
                 float* __restrict__ S1, float* __restrict__ S2) // [KSPLIT][B][256][256]
{
    const int z  = blockIdx.z;
    const int b  = z >> 1;
    const int ks = z & 1;
    const size_t abase = (size_t)b * Cn * Nn;
    const int m0 = blockIdx.y * 64;
    const int n0 = blockIdx.x * 64;
    const int w    = threadIdx.x >> 5;
    const int wm   = w >> 1;          // 0..1 -> 32 rows
    const int wn   = w & 1;           // 0..1 -> 32 cols
    const int lane = threadIdx.x & 31;
    const int sel  = lane >> 4;
    const int r    = lane & 15;

    v8f acc1[2][2] = {}; v8f acc2[2][2] = {};

    const bf16_t* Arow[2];
#pragma unroll
    for (int i = 0; i < 2; ++i)
        Arow[i] = Asum + abase + (size_t)(m0 + wm * 32 + i * 16 + r) * Nn + sel * 16;
    const bf16_t* B1row[2]; const bf16_t* B2row[2];
#pragma unroll
    for (int j = 0; j < 2; ++j) {
        const size_t o = abase + (size_t)(n0 + wn * 32 + j * 16 + r) * Nn + sel * 16;
        B1row[j] = Bt1 + o; B2row[j] = Bt2 + o;
    }

    const int kbeg = ks * (Nn / KSPLIT);
    const int kend = kbeg + (Nn / KSPLIT);
    for (int k0 = kbeg; k0 < kend; k0 += 32) {
        v16bf a[2], b1[2], b2[2];
#pragma unroll
        for (int i = 0; i < 2; ++i) a[i] = load_frag2(Arow[i] + k0, 8);
#pragma unroll
        for (int j = 0; j < 2; ++j) {
            b1[j] = load_frag2(B1row[j] + k0, 8);
            b2[j] = load_frag2(B2row[j] + k0, 8);
        }
#pragma unroll
        for (int i = 0; i < 2; ++i)
#pragma unroll
            for (int j = 0; j < 2; ++j) {
                acc1[i][j] = __builtin_amdgcn_wmma_f32_16x16x32_bf16(
                    false, a[i], false, b1[j], (short)0, acc1[i][j], false, false);
                acc2[i][j] = __builtin_amdgcn_wmma_f32_16x16x32_bf16(
                    false, a[i], false, b2[j], (short)0, acc2[i][j], false, false);
            }
    }

    const size_t obase = ((size_t)ks * Bn + b) * Cn * Cn;
#pragma unroll
    for (int i = 0; i < 2; ++i)
#pragma unroll
        for (int j = 0; j < 2; ++j)
#pragma unroll
            for (int e = 0; e < 8; ++e) {
                const int m = m0 + wm * 32 + i * 16 + sel * 8 + e;
                const int n = n0 + wn * 32 + j * 16 + r;
                S1[obase + (size_t)m * Cn + n] = acc1[i][j][e];
                S2[obase + (size_t)m * Cn + n] = acc2[i][j][e];
            }
}

// ---------------------------------------------------------------------------
// K4: fold split-K partials + row softmax over 256 columns, fp32 in -> bf16 out
// grid (Cn, Bn, 2), block 256
// ---------------------------------------------------------------------------
__global__ void __launch_bounds__(256)
softmax_kernel(const float* __restrict__ S1, const float* __restrict__ S2,
               bf16_t* __restrict__ P1, bf16_t* __restrict__ P2)
{
    const float* S = blockIdx.z ? S2 : S1;
    bf16_t*      P = blockIdx.z ? P2 : P1;
    const size_t row = (size_t)(blockIdx.y * Cn + blockIdx.x) * Cn;
    const size_t part = (size_t)Bn * Cn * Cn;
    const int t = threadIdx.x;
    const float v = S[row + t] + S[part + row + t];   // fold the two K-partials
    __shared__ float red[256];
    red[t] = v; __syncthreads();
    for (int off = 128; off > 0; off >>= 1) {
        if (t < off) red[t] = fmaxf(red[t], red[t + off]);
        __syncthreads();
    }
    const float mx = red[0];
    __syncthreads();
    const float e = __expf(v - mx);
    red[t] = e; __syncthreads();
    for (int off = 128; off > 0; off >>= 1) {
        if (t < off) red[t] += red[t + off];
        __syncthreads();
    }
    P[row + t] = (bf16_t)(e / red[0]);
}

// ---------------------------------------------------------------------------
// K5: back-projection, transposed on the fly:
//   out[b][n][c] = sum_d xT1[n][d]*P1[c][d] + sum_d xT2[n][d]*P2[c][d]
//                  + x1.flat[n*C+c] + x2.flat[n*C+c]
// i.e. yT = xT . P^T with the WMMA B operand being row-major P directly.
// M=4096 (n), N=256 (c), K=256 (d); two streams into ONE accumulator.
// grid (Nn/128, Cn/128, Bn) = 1024 blocks, block 256 (8 waves, 2x4).
// ---------------------------------------------------------------------------
__global__ void __launch_bounds__(256)
out_gemm_kernel(const bf16_t* __restrict__ xT1, const bf16_t* __restrict__ xT2, // [B][4096][256]
                const bf16_t* __restrict__ P1,  const bf16_t* __restrict__ P2,  // [B][256][256]
                const float* __restrict__ x1, const float* __restrict__ x2,
                float* __restrict__ out)
{
    const int b  = blockIdx.z;
    const size_t tbase = (size_t)b * Nn * Cn;   // also the flat base of x1/x2/out
    const size_t pbase = (size_t)b * Cn * Cn;
    const int m0 = blockIdx.x * 128;            // n dimension
    const int n0 = blockIdx.y * 128;            // c dimension
    const int w    = threadIdx.x >> 5;
    const int wm   = w >> 2;
    const int wn   = w & 3;
    const int lane = threadIdx.x & 31;
    const int sel  = lane >> 4;
    const int r    = lane & 15;

    v8f acc[4][2] = {};

    const bf16_t* A1row[4]; const bf16_t* A2row[4];
#pragma unroll
    for (int i = 0; i < 4; ++i) {
        const size_t o = tbase + (size_t)(m0 + wm * 64 + i * 16 + r) * Cn + sel * 16;
        A1row[i] = xT1 + o; A2row[i] = xT2 + o;
    }
    const bf16_t* B1row[2]; const bf16_t* B2row[2];
#pragma unroll
    for (int j = 0; j < 2; ++j) {
        const size_t o = pbase + (size_t)(n0 + wn * 32 + j * 16 + r) * Cn + sel * 16;
        B1row[j] = P1 + o; B2row[j] = P2 + o;
    }

#pragma unroll
    for (int k0 = 0; k0 < Cn; k0 += 32) {
        v16bf a1[4], a2[4], b1[2], b2[2];
#pragma unroll
        for (int i = 0; i < 4; ++i) {
            a1[i] = load_frag2(A1row[i] + k0, 8);
            a2[i] = load_frag2(A2row[i] + k0, 8);
        }
#pragma unroll
        for (int j = 0; j < 2; ++j) {
            b1[j] = load_frag2(B1row[j] + k0, 8);
            b2[j] = load_frag2(B2row[j] + k0, 8);
        }
#pragma unroll
        for (int i = 0; i < 4; ++i)
#pragma unroll
            for (int j = 0; j < 2; ++j) {
                acc[i][j] = __builtin_amdgcn_wmma_f32_16x16x32_bf16(
                    false, a1[i], false, b1[j], (short)0, acc[i][j], false, false);
                acc[i][j] = __builtin_amdgcn_wmma_f32_16x16x32_bf16(
                    false, a2[i], false, b2[j], (short)0, acc[i][j], false, false);
            }
    }

#pragma unroll
    for (int i = 0; i < 4; ++i)
#pragma unroll
        for (int j = 0; j < 2; ++j)
#pragma unroll
            for (int e = 0; e < 8; ++e) {
                const int m = m0 + wm * 64 + i * 16 + sel * 8 + e;
                const int n = n0 + wn * 32 + j * 16 + r;
                const size_t idx = tbase + (size_t)m * Cn + n;
                out[idx] = acc[i][j][e] + x1[idx] + x2[idx];
            }
}

// ---------------------------------------------------------------------------
extern "C" void kernel_launch(void* const* d_in, const int* in_sizes, int n_in,
                              void* d_out, int out_size, void* d_ws, size_t ws_size,
                              hipStream_t stream)
{
    const float* x1 = (const float*)d_in[0];
    const float* x2 = (const float*)d_in[1];
    float* out = (float*)d_out;

    char* ws = (char*)d_ws;
    size_t p = 0;
    auto carve = [&](size_t bytes) -> void* {
        void* q = ws + p;
        p += (bytes + 255) & ~(size_t)255;
        return q;
    };
    const size_t bcn = (size_t)Bn * Cn * Nn;   // 16*256*4096
    bf16_t* x1n = (bf16_t*)carve(bcn * sizeof(bf16_t));
    bf16_t* x2n = (bf16_t*)carve(bcn * sizeof(bf16_t));
    bf16_t* sab = (bf16_t*)carve(bcn * sizeof(bf16_t));
    bf16_t* xT1 = (bf16_t*)carve(bcn * sizeof(bf16_t));
    bf16_t* xT2 = (bf16_t*)carve(bcn * sizeof(bf16_t));
    const size_t bcc = (size_t)Bn * Cn * Cn;
    float*  S1  = (float*)carve(KSPLIT * bcc * sizeof(float));
    float*  S2  = (float*)carve(KSPLIT * bcc * sizeof(float));
    bf16_t* P1  = (bf16_t*)carve(bcc * sizeof(bf16_t));
    bf16_t* P2  = (bf16_t*)carve(bcc * sizeof(bf16_t));

    norm_rows_kernel<<<dim3(Cn, Bn), 256, 0, stream>>>(x1, x2, x1n, x2n, sab);
    transpose_bf16_kernel<<<dim3(Nn / 32, Cn / 32, 2 * Bn), dim3(32, 8), 0, stream>>>(
        x1, x2, xT1, xT2);
    attn_gemm_kernel<<<dim3(Cn / 64, Cn / 64, Bn * KSPLIT), 128, 0, stream>>>(
        sab, x1n, x2n, S1, S2);
    softmax_kernel<<<dim3(Cn, Bn, 2), 256, 0, stream>>>(S1, S2, P1, P2);
    out_gemm_kernel<<<dim3(Nn / 128, Cn / 128, Bn), 256, 0, stream>>>(
        xT1, xT2, P1, P2, x1, x2, out);
}
